// WaveDecoder_62740882260079
// MI455X (gfx1250) — matrix operand order repaired and checked by reading, compiled, not run on hardware
//
#include <hip/hip_runtime.h>
#include <math.h>

typedef __attribute__((ext_vector_type(16))) __bf16 v16bf;
typedef __attribute__((ext_vector_type(8)))  float  v8f;
typedef __attribute__((ext_vector_type(4)))  float  v4f;

#define NLAY 24
#define BATCH 1024
#define RC 64
#define LQ 128
#define HID 256
#define KDIM 128          // concat [prev(64) | h(64)]
#define NGATE 256         // 4*RC
#define SKIPW (NLAY*RC)   // 1536

__device__ __forceinline__ float sigm(float x) { return 1.0f / (1.0f + __expf(-x)); }

// ---------------- weight conversion (f32 -> bf16) ----------------
// wcomb[i][n][k], k<64 -> conv_w[i,n,k,0], k>=64 -> conv_w[i,n,k-64,1]
__global__ void convert_conv_w(const float* __restrict__ conv_w, __bf16* __restrict__ wcomb) {
    int idx = blockIdx.x * 256 + threadIdx.x;          // NLAY*256*128 = 786432
    if (idx >= NLAY * NGATE * KDIM) return;
    int k = idx & 127;
    int n = (idx >> 7) & 255;
    int i = idx >> 15;
    int r = k & 63, s = k >> 6;
    wcomb[idx] = (__bf16)conv_w[(((i * NGATE) + n) * RC + r) * 2 + s];
}

__global__ void convert_fc1_w(const float* __restrict__ fc1_w, __bf16* __restrict__ w) {
    int idx = blockIdx.x * 256 + threadIdx.x;          // HID*SKIPW = 393216
    if (idx >= HID * SKIPW) return;
    w[idx] = (__bf16)fc1_w[idx];
}

// ---------------- bulk queue copy (HBM-roofline part, ~1.6 GB stream) ------
// in:  (NL,B,RC,128) ; out: (NL,B,RC,129), column 128 written by wave_layers.
// Non-temporal (TH=NT) so the 1.6GB stream doesn't evict the WMMA working set
// (bf16 weights / skips) from the 192MB L2.
__global__ __launch_bounds__(256) void copy_queues(const float* __restrict__ q,
                                                   float* __restrict__ outq) {
    size_t tid  = (size_t)blockIdx.x * blockDim.x + threadIdx.x;
    size_t base = tid * 8;                           // 2 x float4 per thread
    if (base >= (size_t)NLAY * BATCH * RC * LQ) return;
#pragma unroll
    for (int p = 0; p < 2; ++p) {
        size_t e   = base + (size_t)p * 4;
        size_t row = e >> 7;
        int    l   = (int)(e & 127);
        v4f v = __builtin_nontemporal_load((const v4f*)(q + e));
        float* o = outq + row * (size_t)(LQ + 1) + l;
        __builtin_nontemporal_store(v.x, o + 0);
        __builtin_nontemporal_store(v.y, o + 1);
        __builtin_nontemporal_store(v.z, o + 2);
        __builtin_nontemporal_store(v.w, o + 3);
    }
}

// ---------------- recurrent 24-layer chain, WMMA bf16 ----------------
__global__ __launch_bounds__(256) void wave_layers(
    const float* __restrict__ x, const float* __restrict__ feat,
    const float* __restrict__ queues,
    const float* __restrict__ fc_h_w, const float* __restrict__ fc_h_b,
    const float* __restrict__ fc_c_w, const float* __restrict__ fc_c_b,
    const __bf16* __restrict__ wcomb, const float* __restrict__ conv_b,
    float* __restrict__ out_q,       // d_out + 1024 (queue region)
    float* __restrict__ skips)       // ws: (B, 1536)
{
    __shared__ float sA[16][KDIM];       // [m][k] = [prev | h]
    __shared__ float hbuf[16][RC];
    __shared__ float cbuf[16][RC];
    __shared__ float cobuf[16][NGATE];

    const int t = threadIdx.x;
    const int lane = t & 31, wave = t >> 5;
    const int b0 = blockIdx.x * 16;
    const int mrow = lane & 15, hf = lane >> 4;

    // ---- initial h, c (K=33 scalar dot, trivial cost) ----
    for (int q = 0; q < 4; ++q) {
        int cell = q * 256 + t;
        int m = cell >> 6, r = cell & 63;
        int b = b0 + m;
        float ah = fc_h_b[r], ac = fc_c_b[r];
        float xi = x[b];
        ah += xi * fc_h_w[r * 33];
        ac += xi * fc_c_w[r * 33];
        for (int j = 0; j < 32; ++j) {
            float f = feat[b * 32 + j];
            ah += f * fc_h_w[r * 33 + 1 + j];
            ac += f * fc_c_w[r * 33 + 1 + j];
        }
        hbuf[m][r] = tanhf(ah);
        cbuf[m][r] = tanhf(ac);
    }
    __syncthreads();

    for (int li = 0; li < NLAY; ++li) {
        const int d = 1 << (li & 7);

        // stage A = [prev | h]; write entry column h into out queue slot 128
        for (int q = 0; q < 4; ++q) {
            int cell = q * 256 + t;
            int m = cell >> 6, r = cell & 63;
            size_t qrow = ((size_t)(li * BATCH + b0 + m)) * RC + r;
            float pv = queues[qrow * LQ + (LQ - d)];
            float hv = hbuf[m][r];
            sA[m][r] = pv;
            sA[m][RC + r] = hv;
            out_q[qrow * (size_t)(LQ + 1) + LQ] = hv;
        }
        __syncthreads();

        // GEMM 16 x 256 x 128; wave owns N in [wave*32, wave*32+32)
        v8f acc0 = {}; v8f acc1 = {};
        for (int kb = 0; kb < 4; ++kb) {
            v16bf a;
#pragma unroll
            for (int j = 0; j < 8; ++j) {
                int k = kb * 32 + (j >> 2) * 16 + hf * 8 + (j & 3) * 2;
                a[2 * j]     = (__bf16)sA[mrow][k];
                a[2 * j + 1] = (__bf16)sA[mrow][k + 1];
            }
#pragma unroll
            for (int nt = 0; nt < 2; ++nt) {
                int n = wave * 32 + nt * 16 + mrow;
                const __bf16* wp = wcomb + ((size_t)(li * NGATE + n)) * KDIM + kb * 32 + hf * 8;
                v16bf bm;
#pragma unroll
                for (int j = 0; j < 16; ++j) bm[j] = wp[j < 8 ? j : j + 8];
                if (nt == 0)
                    acc0 = __builtin_amdgcn_wmma_f32_16x16x32_bf16(false, a, false, bm, (short)0, acc0, false, false);
                else
                    acc1 = __builtin_amdgcn_wmma_f32_16x16x32_bf16(false, a, false, bm, (short)0, acc1, false, false);
            }
        }
        // scatter C (16x16 f32 layout: lane = N, vgpr v -> M = v + 8*half)
#pragma unroll
        for (int nt = 0; nt < 2; ++nt) {
            v8f acc = nt ? acc1 : acc0;
            int n = wave * 32 + nt * 16 + mrow;
            float bias = conv_b[li * NGATE + n];
#pragma unroll
            for (int v = 0; v < 8; ++v)
                cobuf[v + 8 * hf][n] = acc[v] + bias;
        }
        __syncthreads();

        // gate math + skip write
        for (int q = 0; q < 4; ++q) {
            int cell = q * 256 + t;
            int m = cell >> 6, r = cell & 63;
            float ig = cobuf[m][r];
            float cf = cobuf[m][RC + r];
            float cg = cobuf[m][2 * RC + r];
            float eg = cobuf[m][3 * RC + r];
            float cv = sigm(ig) * cbuf[m][r] + tanhf(cf) * sigm(cg);
            float hv = sigm(eg) * tanhf(cv);
            cbuf[m][r] = cv;
            hbuf[m][r] = hv;
            skips[(size_t)(b0 + m) * SKIPW + li * RC + r] = hv;
        }
        __syncthreads();
    }
}

// ---------------- fc1 (WMMA, K=1536) + relu + fc2 ----------------
__global__ __launch_bounds__(256) void fc_head(
    const float* __restrict__ skips, const __bf16* __restrict__ w1,
    const float* __restrict__ fc1_b,
    const float* __restrict__ fc2_w, const float* __restrict__ fc2_b,
    float* __restrict__ hhat)
{
    __shared__ float ybuf[16][HID];
    const int t = threadIdx.x;
    const int lane = t & 31, wave = t >> 5;
    const int b0 = blockIdx.x * 16;
    const int mrow = lane & 15, hf = lane >> 4;

    v8f acc0 = {}; v8f acc1 = {};
    for (int kb = 0; kb < SKIPW / 32; ++kb) {   // 48 k-blocks
        v16bf a;
        const float* ap = skips + (size_t)(b0 + mrow) * SKIPW + kb * 32 + hf * 8;
#pragma unroll
        for (int j = 0; j < 8; ++j) {
            int k = (j >> 2) * 16 + (j & 3) * 2;
            a[2 * j]     = (__bf16)ap[k];
            a[2 * j + 1] = (__bf16)ap[k + 1];
        }
#pragma unroll
        for (int nt = 0; nt < 2; ++nt) {
            int n = wave * 32 + nt * 16 + mrow;
            const __bf16* wp = w1 + (size_t)n * SKIPW + kb * 32 + hf * 8;
            v16bf bm;
#pragma unroll
            for (int j = 0; j < 16; ++j) bm[j] = wp[j < 8 ? j : j + 8];
            if (nt == 0)
                acc0 = __builtin_amdgcn_wmma_f32_16x16x32_bf16(false, a, false, bm, (short)0, acc0, false, false);
            else
                acc1 = __builtin_amdgcn_wmma_f32_16x16x32_bf16(false, a, false, bm, (short)0, acc1, false, false);
        }
    }
#pragma unroll
    for (int nt = 0; nt < 2; ++nt) {
        v8f acc = nt ? acc1 : acc0;
        int n = wave * 32 + nt * 16 + mrow;
        float bias = fc1_b[n];
#pragma unroll
        for (int v = 0; v < 8; ++v) {
            float y = acc[v] + bias;
            ybuf[v + 8 * hf][n] = y > 0.0f ? y : 0.0f;
        }
    }
    __syncthreads();
    if (t < 16) {
        float s = fc2_b[0];
        for (int j = 0; j < HID; ++j) s += ybuf[t][j] * fc2_w[j];
        hhat[b0 + t] = s;
    }
}

extern "C" void kernel_launch(void* const* d_in, const int* in_sizes, int n_in,
                              void* d_out, int out_size, void* d_ws, size_t ws_size,
                              hipStream_t stream) {
    (void)in_sizes; (void)n_in; (void)out_size; (void)ws_size;
    const float* x      = (const float*)d_in[0];
    const float* feat   = (const float*)d_in[1];
    const float* queues = (const float*)d_in[2];
    const float* fc_h_w = (const float*)d_in[3];
    const float* fc_h_b = (const float*)d_in[4];
    const float* fc_c_w = (const float*)d_in[5];
    const float* fc_c_b = (const float*)d_in[6];
    const float* conv_w = (const float*)d_in[7];
    const float* conv_b = (const float*)d_in[8];
    const float* fc1_w  = (const float*)d_in[9];
    const float* fc1_b  = (const float*)d_in[10];
    const float* fc2_w  = (const float*)d_in[11];
    const float* fc2_b  = (const float*)d_in[12];

    float* out  = (float*)d_out;
    float* hhat = out;                 // (B,1,1) = 1024 floats
    float* outq = out + BATCH;         // (NL,B,RC,129)

    char* ws = (char*)d_ws;
    __bf16* wcomb = (__bf16*)ws;                                   // 786432 bf16
    __bf16* w1bf  = (__bf16*)(ws + (size_t)786432 * 2);            // 393216 bf16
    float*  skips = (float*)(ws + (size_t)786432 * 2 + 393216 * 2);// 1024*1536 f32

    hipLaunchKernelGGL(convert_conv_w, dim3((NLAY * NGATE * KDIM + 255) / 256), dim3(256), 0, stream,
                       conv_w, wcomb);
    hipLaunchKernelGGL(convert_fc1_w, dim3((HID * SKIPW + 255) / 256), dim3(256), 0, stream,
                       fc1_w, w1bf);

    size_t total = (size_t)NLAY * BATCH * RC * LQ;   // 201326592
    size_t nthr  = total / 8;                        // 25165824 (8 floats/thread)
    hipLaunchKernelGGL(copy_queues, dim3((unsigned)(nthr / 256)), dim3(256), 0, stream,
                       queues, outq);

    hipLaunchKernelGGL(wave_layers, dim3(BATCH / 16), dim3(256), 0, stream,
                       x, feat, queues, fc_h_w, fc_h_b, fc_c_w, fc_c_b,
                       wcomb, conv_b, outq, skips);

    hipLaunchKernelGGL(fc_head, dim3(BATCH / 16), dim3(256), 0, stream,
                       skips, w1bf, fc1_b, fc2_w, fc2_b, hhat);
}
